// MoELayer_23493471109263
// MI455X (gfx1250) — matrix operand order repaired and checked by reading, not compiled, not run on hardware
//
#include <hip/hip_runtime.h>
#include <hip/hip_bf16.h>

// ---------------- problem constants (match reference) ----------------
constexpr int cD = 1024;   // hidden dim
constexpr int cH = 4096;   // expert ffn dim
constexpr int cE = 8;      // experts
constexpr int cT = 2048;   // tokens = B*S

typedef __attribute__((ext_vector_type(16))) __bf16 v16bf;
typedef __attribute__((ext_vector_type(8)))  __bf16 v8bf;
typedef __attribute__((ext_vector_type(4)))  __bf16 v4bf;
typedef __attribute__((ext_vector_type(8)))  float  v8f;

// ---------------- fragment loaders ----------------
// A-matrix 16x32 bf16 per-lane pattern: lane holds row m=lane&15,
// k = 16*half+0..15 (contiguous, matches B), half = lane>>4.  Two 16B loads.
__device__ __forceinline__ v16bf frag_a_bf16(const __bf16* row, int k0, int half) {
  int off = k0 + (half << 4);
  v8bf r0 = *(const v8bf*)(row + off);
  v8bf r1 = *(const v8bf*)(row + off + 8);
  return __builtin_shufflevector(r0, r1, 0,1,2,3,4,5,6,7,8,9,10,11,12,13,14,15);
}
// B-matrix 32x16 bf16 per-lane pattern (same contiguous k order as A): lane holds
// column n=lane&15, 16 consecutive k at k0 + 16*half.  Source is row-major f32
// (weight rows), converted to bf16 on the fly (64B contiguous load).
__device__ __forceinline__ v16bf frag_b_f32(const float* row, int k0, int half) {
  const float4* p = (const float4*)(row + k0 + (half << 4));
  float4 f0 = p[0], f1 = p[1], f2 = p[2], f3 = p[3];
  v16bf r;
  r[0]=(__bf16)f0.x;  r[1]=(__bf16)f0.y;  r[2]=(__bf16)f0.z;  r[3]=(__bf16)f0.w;
  r[4]=(__bf16)f1.x;  r[5]=(__bf16)f1.y;  r[6]=(__bf16)f1.z;  r[7]=(__bf16)f1.w;
  r[8]=(__bf16)f2.x;  r[9]=(__bf16)f2.y;  r[10]=(__bf16)f2.z; r[11]=(__bf16)f2.w;
  r[12]=(__bf16)f3.x; r[13]=(__bf16)f3.y; r[14]=(__bf16)f3.z; r[15]=(__bf16)f3.w;
  return r;
}

// ---------------- K0: zero output + expert counters ----------------
__global__ void moe_zero_kernel(float* __restrict__ out, int* __restrict__ counts) {
  size_t i = (size_t)blockIdx.x * blockDim.x + threadIdx.x;
  size_t N = (size_t)cT * cD + 10;
  if (i < N) out[i] = 0.0f;
  if (blockIdx.x == 0 && threadIdx.x < cE) counts[threadIdx.x] = 0;
}

// ---------------- K1: x (f32) -> bf16 copy ----------------
__global__ void moe_cvt_kernel(const float* __restrict__ x, __bf16* __restrict__ xb) {
  int i = (blockIdx.x * 256 + threadIdx.x) * 4;
  float4 v = *(const float4*)(x + i);
  v4bf o;
  o[0] = (__bf16)v.x; o[1] = (__bf16)v.y; o[2] = (__bf16)v.z; o[3] = (__bf16)v.w;
  *(v4bf*)(xb + i) = o;
}

// ---------------- K2: router (one wave32 per token) ----------------
__global__ void moe_router_kernel(const float* __restrict__ x,
                                  const float* __restrict__ rw,
                                  float* __restrict__ probs,
                                  float* __restrict__ lse2,
                                  float* __restrict__ combine,
                                  int* __restrict__ counts,
                                  int* __restrict__ lists) {
  int wave = threadIdx.x >> 5;
  int lane = threadIdx.x & 31;
  int t = blockIdx.x * 8 + wave;

  float acc[cE];
#pragma unroll
  for (int e = 0; e < cE; ++e) acc[e] = 0.0f;

  const float* xr = x + (size_t)t * cD;
  for (int d = lane; d < cD; d += 32) {
    float xv = xr[d];
#pragma unroll
    for (int e = 0; e < cE; ++e) acc[e] += xv * rw[e * cD + d];
  }
  // wave32 butterfly reduce (fixed order -> deterministic)
#pragma unroll
  for (int e = 0; e < cE; ++e) {
#pragma unroll
    for (int m = 16; m >= 1; m >>= 1) acc[e] += __shfl_xor(acc[e], m, 32);
  }

  // softmax over all E
  float mx = acc[0];
#pragma unroll
  for (int e = 1; e < cE; ++e) mx = fmaxf(mx, acc[e]);
  float ex[cE]; float se = 0.0f;
#pragma unroll
  for (int e = 0; e < cE; ++e) { ex[e] = __expf(acc[e] - mx); se += ex[e]; }
  float inv = 1.0f / se;
  float lse = mx + __logf(se);

  // top-2 (first occurrence on ties, matching lax.top_k)
  int i0 = 0;
#pragma unroll
  for (int e = 1; e < cE; ++e) if (acc[e] > acc[i0]) i0 = e;
  int i1 = -1;
#pragma unroll
  for (int e = 0; e < cE; ++e) if (e != i0 && (i1 < 0 || acc[e] > acc[i1])) i1 = e;
  float eg = __expf(acc[i1] - acc[i0]);
  float w0 = 1.0f / (1.0f + eg);
  float w1 = eg * w0;

  if (lane == 0) {
#pragma unroll
    for (int e = 0; e < cE; ++e) probs[t * cE + e] = ex[e] * inv;
    lse2[t] = lse * lse;
#pragma unroll
    for (int e = 0; e < cE; ++e)
      combine[t * cE + e] = (e == i0) ? w0 : ((e == i1) ? w1 : 0.0f);
    int p0 = atomicAdd(&counts[i0], 1); lists[i0 * cT + p0] = t;
    int p1 = atomicAdd(&counts[i1], 1); lists[i1 * cT + p1] = t;
  }
}

// ---------------- K3: deterministic loss / usage reductions ----------------
__global__ void moe_losses_kernel(const float* __restrict__ probs,
                                  const float* __restrict__ lse2,
                                  float* __restrict__ tail) {
  __shared__ float red[256];
  int tid = threadIdx.x;
  float usage[cE];
  for (int e = 0; e < cE; ++e) {
    float s = 0.0f;
    for (int t = tid; t < cT; t += 256) s += probs[t * cE + e];
    red[tid] = s; __syncthreads();
    for (int st = 128; st >= 1; st >>= 1) {
      if (tid < st) red[tid] += red[tid + st];
      __syncthreads();
    }
    usage[e] = red[0] * (1.0f / cT);
    __syncthreads();
  }
  float z = 0.0f;
  for (int t = tid; t < cT; t += 256) z += lse2[t];
  red[tid] = z; __syncthreads();
  for (int st = 128; st >= 1; st >>= 1) {
    if (tid < st) red[tid] += red[tid + st];
    __syncthreads();
  }
  if (tid == 0) {
    float lb = 0.0f;
    for (int e = 0; e < cE; ++e) lb += usage[e] * usage[e];
    tail[0] = lb * (float)cE;          // load_balance_loss
    tail[1] = red[0] * (1.0f / cT);    // router_z_loss
    for (int e = 0; e < cE; ++e) tail[2 + e] = usage[e];
  }
}

// ---------------- K4: fused expert FFN (compacted tokens, bf16 WMMA) --------
// block = (expert e, tile of 16 tokens); 8 waves.
// Per 128-wide H chunk:  phase1: wave w computes gate/up tile for h-cols
// [hb+16w, hb+16w+16) over K=D (64 WMMA), SiLU*mul -> bf16 Ps in LDS.
// phase2: wave w accumulates out cols [128w,128w+128) over K=128 (32 WMMA)
// into persistent v8f accumulators.  Final combine-weighted atomic scatter.
__global__ __launch_bounds__(256) void moe_ffn_kernel(
    const __bf16* __restrict__ xb, const float* __restrict__ w1,
    const float* __restrict__ w2, const float* __restrict__ w3,
    const float* __restrict__ combine, const int* __restrict__ counts,
    const int* __restrict__ lists, float* __restrict__ out) {
  int e    = blockIdx.x >> 7;    // 128 tiles per expert (worst case)
  int tile = blockIdx.x & 127;
  int cnt  = counts[e];
  int m0   = tile * 16;
  if (m0 >= cnt) return;         // uniform per block

  __shared__ __bf16 Xs[16][1032];   // +16B/row pad: conflict-free A-frag reads
  __shared__ __bf16 Ps[16][136];
  __shared__ int    tokS[16];
  __shared__ float  cwS[16];

  int tid = threadIdx.x;
  if (tid < 16) {
    int idx = m0 + tid;
    if (idx < cnt) {
      int t = lists[e * cT + idx];
      tokS[tid] = t;
      cwS[tid]  = combine[t * cE + e];
    } else {
      tokS[tid] = -1;
      cwS[tid]  = 0.0f;
    }
  }
  __syncthreads();

  // gather 16 token rows of bf16 x into LDS (16B per iteration)
  for (int i = tid; i < 16 * 128; i += 256) {
    int r  = i >> 7;
    int c8 = (i & 127) << 3;
    int4 v = make_int4(0, 0, 0, 0);
    int t = tokS[r];
    if (t >= 0) v = *(const int4*)(xb + (size_t)t * cD + c8);
    *(int4*)(&Xs[r][c8]) = v;
  }
  __syncthreads();

  int wave = tid >> 5, lane = tid & 31;
  int half = lane >> 4, nl = lane & 15;
  int dbase = wave << 7;

  v8f zz = {0.f, 0.f, 0.f, 0.f, 0.f, 0.f, 0.f, 0.f};
  v8f oacc[8];
#pragma unroll
  for (int n = 0; n < 8; ++n) oacc[n] = zz;

  const float* w1e = w1 + (size_t)e * cH * cD;
  const float* w3e = w3 + (size_t)e * cH * cD;
  const float* w2e = w2 + (size_t)e * cD * cH;

  for (int hb = 0; hb < cH; hb += 128) {
    // ---- phase 1: gate/up for this wave's 16 h-columns ----
    int hc = hb + (wave << 4) + nl;
    const float* w1row = w1e + (size_t)hc * cD;
    const float* w3row = w3e + (size_t)hc * cD;
    const __bf16* xrow = &Xs[nl][0];
    v8f g = zz, u = zz;
#pragma unroll 4
    for (int k0 = 0; k0 < cD; k0 += 32) {
      __builtin_prefetch(w1row + k0 + 256, 0, 1);
      __builtin_prefetch(w3row + k0 + 256, 0, 1);
      v16bf a  = frag_a_bf16(xrow, k0, half);
      v16bf b1 = frag_b_f32(w1row, k0, half);
      v16bf b3 = frag_b_f32(w3row, k0, half);
      g = __builtin_amdgcn_wmma_f32_16x16x32_bf16(false, a, false, b1, (short)0, g, false, false);
      u = __builtin_amdgcn_wmma_f32_16x16x32_bf16(false, a, false, b3, (short)0, u, false, false);
    }
    // silu(g)*u -> bf16 product tile (C-layout: row m=j+8*half, col=16*wave+nl)
#pragma unroll
    for (int j = 0; j < 8; ++j) {
      float gv = g[j];
      float pv = (gv / (1.0f + __expf(-gv))) * u[j];
      Ps[j + (half << 3)][(wave << 4) + nl] = (__bf16)pv;
    }
    __syncthreads();

    // ---- phase 2: out[16, dbase:dbase+128] += P[16,128] @ w2_chunk^T ----
#pragma unroll
    for (int kk = 0; kk < 128; kk += 32) {
      v16bf ap = frag_a_bf16(&Ps[nl][0], kk, half);
#pragma unroll
      for (int n = 0; n < 8; ++n) {
        int dc = dbase + (n << 4) + nl;
        const float* w2row = w2e + (size_t)dc * cH + hb;
        v16bf b2 = frag_b_f32(w2row, kk, half);
        oacc[n] = __builtin_amdgcn_wmma_f32_16x16x32_bf16(false, ap, false, b2, (short)0, oacc[n], false, false);
      }
    }
    __syncthreads();   // Ps is rewritten next chunk
  }

  // combine-weighted scatter; each out element gets exactly 2 commutative adds
#pragma unroll
  for (int j = 0; j < 8; ++j) {
    int m = j + (half << 3);
    int t = tokS[m];
    if (t < 0) continue;
    float cw = cwS[m];
#pragma unroll
    for (int n = 0; n < 8; ++n) {
      atomicAdd(out + (size_t)t * cD + dbase + (n << 4) + nl, cw * oacc[n][j]);
    }
  }
}

// ---------------- launch ----------------
extern "C" void kernel_launch(void* const* d_in, const int* in_sizes, int n_in,
                              void* d_out, int out_size, void* d_ws, size_t ws_size,
                              hipStream_t stream) {
  (void)in_sizes; (void)n_in; (void)out_size; (void)ws_size;
  const float* x  = (const float*)d_in[0];
  const float* rw = (const float*)d_in[1];
  const float* w1 = (const float*)d_in[2];
  const float* w2 = (const float*)d_in[3];
  const float* w3 = (const float*)d_in[4];
  float* out = (float*)d_out;

  // workspace layout (~4.4 MB)
  char* ws = (char*)d_ws;
  __bf16* xb     = (__bf16*)ws;                       size_t off = (size_t)cT * cD * 2;
  float* combine = (float*)(ws + off);                off += (size_t)cT * cE * 4;
  float* probs   = (float*)(ws + off);                off += (size_t)cT * cE * 4;
  float* lse2    = (float*)(ws + off);                off += (size_t)cT * 4;
  int*   counts  = (int*)(ws + off);                  off += 64;
  int*   lists   = (int*)(ws + off);

  size_t nzero = (size_t)cT * cD + 10;
  moe_zero_kernel<<<(int)((nzero + 255) / 256), 256, 0, stream>>>(out, counts);
  moe_cvt_kernel<<<(cT * cD / 4) / 256, 256, 0, stream>>>(x, xb);
  moe_router_kernel<<<cT / 8, 256, 0, stream>>>(x, rw, probs, lse2, combine, counts, lists);
  moe_losses_kernel<<<1, 256, 0, stream>>>(probs, lse2, out + (size_t)cT * cD);
  moe_ffn_kernel<<<cE * (cT / 16), 256, 0, stream>>>(xb, w1, w2, w3, combine, counts, lists, out);
}
